// Feature_RNNLM_49692771615180
// MI455X (gfx1250) — matrix operand order repaired and checked by reading, not compiled, not run on hardware
//
#include <hip/hip_runtime.h>
#include <hip/hip_bf16.h>
#include <math.h>

typedef __attribute__((ext_vector_type(16))) _Float16 v16h;
typedef __attribute__((ext_vector_type(8)))  _Float16 v8h;
typedef __attribute__((ext_vector_type(8)))  float    v8f;

// ---------------------------------------------------------------------------
// WMMA helpers (wave32, V_WMMA_F32_16X16X32_F16 layouts per CDNA5 ISA 7.12.2)
// ---------------------------------------------------------------------------

static __device__ __forceinline__ v8f wmma_f16(v16h a, v16h b, v8f c) {
  // (neg_a, A, neg_b, B, c_mod, C, reuse_a, reuse_b)
  return __builtin_amdgcn_wmma_f32_16x16x32_f16(false, a, false, b, (short)0, c,
                                                false, false);
}

static __device__ __forceinline__ float fast_tanh(float x) {
#if __has_builtin(__builtin_amdgcn_tanhf)
  return __builtin_amdgcn_tanhf(x);   // single V_TANH_F32 trans op on gfx1250
#else
  return tanhf(x);
#endif
}

// A fragment: 16x32 (MxK), row-major f16 source, base = start of row 0.
// lane L: row = L%16 ; kbase = (L>=16)?8:0
// halves 0..7  -> K = k0+kbase .. k0+kbase+7   (16B load)
// halves 8..15 -> K = k0+kbase+16 .. +23       (16B load)
static __device__ __forceinline__ v16h load_a_frag(const _Float16* base, int lda,
                                                   int lane, int k0) {
  const int r  = lane & 15;
  const int kb = k0 + ((lane >> 4) << 3);
  const _Float16* p = base + (size_t)r * lda + kb;
  v8h lo = *(const v8h*)(p);
  v8h hi = *(const v8h*)(p + 16);
  v16h o;
#pragma unroll
  for (int i = 0; i < 8; ++i) { o[i] = lo[i]; o[i + 8] = hi[i]; }
  return o;
}

// B fragment: 32x16 (KxN) where B[k][n] = W[n][k], W row-major [N][K] f16.
// lane L: col n = n0 + L%16 (clamped to nmax); kbase = (L>=16)?8:0
// halves 0..7 -> K = k0+kbase..+7 ; halves 8..15 -> K = k0+kbase+16..+23
static __device__ __forceinline__ v16h load_b_frag(const _Float16* W, int ldw,
                                                   int n0, int nmax, int lane,
                                                   int k0) {
  int n = n0 + (lane & 15);
  if (n > nmax) n = nmax;  // clamp (no branch around WMMA)
  const int kb = k0 + ((lane >> 4) << 3);
  const _Float16* p = W + (size_t)n * ldw + kb;
  v8h lo = *(const v8h*)(p);
  v8h hi = *(const v8h*)(p + 16);
  v16h o;
#pragma unroll
  for (int i = 0; i < 8; ++i) { o[i] = lo[i]; o[i + 8] = hi[i]; }
  return o;
}

// ---------------------------------------------------------------------------
// fp32 -> fp16 conversion (weights / embedding table)
// ---------------------------------------------------------------------------
__global__ void cvt_f16_kernel(const float* __restrict__ s,
                               _Float16* __restrict__ d, int n) {
  int i = blockIdx.x * blockDim.x + threadIdx.x;
  if (i < n) d[i] = (_Float16)s[i];
}

// ---------------------------------------------------------------------------
// Embedding gather fused with layer-0 input projection:
//   xp0[r, n] = sum_k feat16[batch[r], k] * W_ih0[n, k] + b_ih0[n] + b_hh0[n]
// r = b*S+s in [0, 32768), H=256, F=64. One wave per 16x16 tile, 8 waves/block.
// ---------------------------------------------------------------------------
__global__ __launch_bounds__(256) void gather_proj0_kernel(
    const int* __restrict__ batch, const _Float16* __restrict__ feat,
    const _Float16* __restrict__ Wih, const float* __restrict__ bi,
    const float* __restrict__ bh, float* __restrict__ xp) {
  const int lane = threadIdx.x & 31;
  const int wave = threadIdx.x >> 5;
  const int tile = blockIdx.x * 8 + wave;   // 2048 m-tiles x 16 n-tiles
  const int tm = tile >> 4, tn = tile & 15;
  const int m0 = tm * 16, n0 = tn * 16;
  const int n = n0 + (lane & 15);

  // gathered A row for this lane
  const int tok = batch[m0 + (lane & 15)];
  const _Float16* arow = feat + (size_t)tok * 64;
  const int kba = (lane >> 4) << 3;

  const float bv = bi[n] + bh[n];
  v8f acc;
#pragma unroll
  for (int v = 0; v < 8; ++v) acc[v] = bv;

#pragma unroll
  for (int c = 0; c < 2; ++c) {  // K = 64 = 2 x 32
    const int k0 = c * 32;
    const _Float16* p = arow + k0 + kba;
    v8h lo = *(const v8h*)(p);
    v8h hi = *(const v8h*)(p + 16);
    v16h a;
#pragma unroll
    for (int i = 0; i < 8; ++i) { a[i] = lo[i]; a[i + 8] = hi[i]; }
    v16h b = load_b_frag(Wih, 64, n0, 255, lane, k0);
    acc = wmma_f16(a, b, acc);
  }

  const int mo = (lane >> 4) << 3;
#pragma unroll
  for (int v = 0; v < 8; ++v)
    xp[(size_t)(m0 + mo + v) * 256 + n] = acc[v];
}

// ---------------------------------------------------------------------------
// Recurrent scan: h_t = tanh(xp_t + h_{t-1} @ W_hh^T), single workgroup.
// 512 threads = 16 waves; wave w owns n-tile w and all 4 m-tiles (B=64).
// W_hh fragments (8 x v16h = 64 VGPRs) held in registers across all 512 steps.
// Hidden state double-buffered in LDS as f16 (2 x 32KB).
// Time loop unrolled by 2: xp for step t+1 is loaded into a second register
// buffer while step t computes, hiding global/L2 latency behind WMMA+barrier.
// S must be even (S = 512 here).
// ---------------------------------------------------------------------------
__global__ __launch_bounds__(512) void rnn_scan_kernel(
    const float* __restrict__ xp,       // [B][S][H] f32 (input part + biases)
    const _Float16* __restrict__ Whh,   // [H][H] f16
    _Float16* __restrict__ h_all,       // [B][S][H] f16 output
    int S) {
  __shared__ __align__(16) _Float16 hbuf0[64 * 256];
  __shared__ __align__(16) _Float16 hbuf1[64 * 256];

  const int lane = threadIdx.x & 31;
  const int wave = threadIdx.x >> 5;    // n-tile id, 0..15
  const int n0 = wave * 16;
  const int n = n0 + (lane & 15);
  const int mo = (lane >> 4) << 3;

  // resident W_hh fragments for this wave's n-tile (full K = 256)
  v16h bfrag[8];
#pragma unroll
  for (int c = 0; c < 8; ++c)
    bfrag[c] = load_b_frag(Whh, 256, n0, 255, lane, c * 32);

  // h_{-1} = 0
  for (int i = threadIdx.x; i < 64 * 256; i += 512) hbuf0[i] = (_Float16)0.0f;
  __syncthreads();

  // xp load: 4 m-tiles x 8 rows for time step t into a register buffer
  float xbufA[4][8], xbufB[4][8];
  auto load_xp = [&](int t, float (&dst)[4][8]) {
#pragma unroll
    for (int mt = 0; mt < 4; ++mt)
#pragma unroll
      for (int v = 0; v < 8; ++v)
        dst[mt][v] = xp[((size_t)(mt * 16 + mo + v) * S + t) * 256 + n];
  };

  // one recurrence step: reads hprev (LDS), writes hnext (LDS) + h_all (global)
  auto step = [&](int t, float (&xv)[4][8], const _Float16* hprev,
                  _Float16* hnext) {
#pragma unroll
    for (int mt = 0; mt < 4; ++mt) {
      const int m0 = mt * 16;
      v8f acc;
#pragma unroll
      for (int v = 0; v < 8; ++v) acc[v] = xv[mt][v];
#pragma unroll
      for (int c = 0; c < 8; ++c) {
        v16h a = load_a_frag(hprev + (size_t)m0 * 256, 256, lane, c * 32);
        acc = wmma_f16(a, bfrag[c], acc);
      }
#pragma unroll
      for (int v = 0; v < 8; ++v) {
        const int m = m0 + mo + v;
        const _Float16 hv = (_Float16)fast_tanh(acc[v]);
        hnext[m * 256 + n] = hv;
        h_all[((size_t)m * S + t) * 256 + n] = hv;
      }
    }
  };

  load_xp(0, xbufA);
  for (int t = 0; t < S; t += 2) {
    // prefetch further ahead into L2 (global_prefetch_b8)
    if (t + 4 < S)
      __builtin_prefetch(&xp[((size_t)(mo)*S + (t + 4)) * 256 + n], 0, 0);

    load_xp(t + 1, xbufB);       // issue loads; consumed after the barrier
    step(t, xbufA, hbuf0, hbuf1);
    __syncthreads();

    if (t + 2 < S) load_xp(t + 2, xbufA);
    step(t + 1, xbufB, hbuf1, hbuf0);
    __syncthreads();
  }
}

// ---------------------------------------------------------------------------
// Generic f16 WMMA GEMM: C[r,n] = sum_k A[r,k]*W[n,k] + bias0[n] (+ bias1[n])
// A: [M][K] f16 row-major, W: [N][K] f16 row-major, C: [M][N] f32.
// One wave per 16x16 output tile, 8 waves per block. N may be non-mult-of-16.
// ---------------------------------------------------------------------------
__global__ __launch_bounds__(256) void gemm16_kernel(
    const _Float16* __restrict__ A, const _Float16* __restrict__ W,
    const float* __restrict__ bias0, const float* __restrict__ bias1,
    float* __restrict__ C, int tiles_n, int N, int K, int total_tiles) {
  const int lane = threadIdx.x & 31;
  const int wave = threadIdx.x >> 5;
  const int tile = blockIdx.x * 8 + wave;
  if (tile >= total_tiles) return;  // wave-uniform guard
  const int tm = tile / tiles_n, tn = tile % tiles_n;
  const int m0 = tm * 16, n0 = tn * 16;
  const int n = n0 + (lane & 15);
  const int nc = n < N ? n : N - 1;

  float bv = bias0[nc];
  if (bias1) bv += bias1[nc];
  v8f acc;
#pragma unroll
  for (int v = 0; v < 8; ++v) acc[v] = bv;

  const _Float16* abase = A + (size_t)m0 * K;
#pragma unroll 8
  for (int k0 = 0; k0 < K; k0 += 32) {
    v16h a = load_a_frag(abase, K, lane, k0);
    v16h b = load_b_frag(W, K, n0, N - 1, lane, k0);
    acc = wmma_f16(a, b, acc);
  }

  const int mo = (lane >> 4) << 3;
  if (n < N) {
#pragma unroll
    for (int v = 0; v < 8; ++v)
      C[(size_t)(m0 + mo + v) * N + n] = acc[v];
  }
}

// ---------------------------------------------------------------------------
// Host launch
// ---------------------------------------------------------------------------
extern "C" void kernel_launch(void* const* d_in, const int* in_sizes, int n_in,
                              void* d_out, int out_size, void* d_ws,
                              size_t ws_size, hipStream_t stream) {
  (void)in_sizes; (void)n_in; (void)out_size; (void)ws_size;

  const int* batch   = (const int*)d_in[0];
  const float* features = (const float*)d_in[1];
  const float* W_ih0 = (const float*)d_in[2];
  const float* W_hh0 = (const float*)d_in[3];
  const float* b_ih0 = (const float*)d_in[4];
  const float* b_hh0 = (const float*)d_in[5];
  const float* W_ih1 = (const float*)d_in[6];
  const float* W_hh1 = (const float*)d_in[7];
  const float* b_ih1 = (const float*)d_in[8];
  const float* b_hh1 = (const float*)d_in[9];
  const float* W_out = (const float*)d_in[10];
  const float* b_out = (const float*)d_in[11];
  float* out = (float*)d_out;

  const int S = 512, H = 256, V = 1000, F = 64;
  const size_t M = (size_t)64 * 512;  // 32768 (b,s) rows

  // workspace carve-out (all 256B aligned), ~102 MB total
  char* ws = (char*)d_ws;
  size_t off = 0;
  auto carve = [&](size_t bytes) -> void* {
    void* p = ws + off;
    off = (off + bytes + 255) & ~(size_t)255;
    return p;
  };
  float*    xp0    = (float*)carve(M * H * sizeof(float));
  float*    xp1    = (float*)carve(M * H * sizeof(float));
  _Float16* h0     = (_Float16*)carve(M * H * sizeof(_Float16));
  _Float16* h1     = (_Float16*)carve(M * H * sizeof(_Float16));
  _Float16* feat16 = (_Float16*)carve((size_t)V * F * sizeof(_Float16));
  _Float16* wih0   = (_Float16*)carve((size_t)H * F * sizeof(_Float16));
  _Float16* whh0   = (_Float16*)carve((size_t)H * H * sizeof(_Float16));
  _Float16* wih1   = (_Float16*)carve((size_t)H * H * sizeof(_Float16));
  _Float16* whh1   = (_Float16*)carve((size_t)H * H * sizeof(_Float16));
  _Float16* wout   = (_Float16*)carve((size_t)V * H * sizeof(_Float16));

  auto cvt = [&](const float* s, _Float16* d, int n) {
    cvt_f16_kernel<<<(n + 255) / 256, 256, 0, stream>>>(s, d, n);
  };
  cvt(features, feat16, V * F);
  cvt(W_ih0, wih0, H * F);
  cvt(W_hh0, whh0, H * H);
  cvt(W_ih1, wih1, H * H);
  cvt(W_hh1, whh1, H * H);
  cvt(W_out, wout, V * H);

  // layer-0 input projection (with embedding gather): 2048*16 tiles / 8 per blk
  gather_proj0_kernel<<<4096, 256, 0, stream>>>(batch, feat16, wih0, b_ih0,
                                                b_hh0, xp0);

  // layer-0 recurrence (sequential; single workgroup)
  rnn_scan_kernel<<<1, 512, 0, stream>>>(xp0, whh0, h0, S);

  // layer-1 input projection: xp1 = h0 @ W_ih1^T + b_ih1 + b_hh1
  {
    const int tiles_n = H / 16;                 // 16
    const int total = (int)(M / 16) * tiles_n;  // 32768
    gemm16_kernel<<<(total + 7) / 8, 256, 0, stream>>>(
        h0, wih1, b_ih1, b_hh1, xp1, tiles_n, H, H, total);
  }

  // layer-1 recurrence
  rnn_scan_kernel<<<1, 512, 0, stream>>>(xp1, whh1, h1, S);

  // output projection: logits = h1 @ W_out^T + b_out  (N = 1000, edge tile)
  {
    const int tiles_n = (V + 15) / 16;          // 63
    const int total = (int)(M / 16) * tiles_n;  // 129024
    gemm16_kernel<<<(total + 7) / 8, 256, 0, stream>>>(
        h1, wout, b_out, nullptr, out, tiles_n, V, H, total);
  }
}
